// EGNNLiteLayer_19868518711570
// MI455X (gfx1250) — matrix operand order repaired and checked by reading, not compiled, not run on hardware
//
#include <hip/hip_runtime.h>
#include <hip/hip_bf16.h>
#include <cstdint>

// ---------------------------------------------------------------------------
// EGNN-lite layer for MI455X (gfx1250).
// Compute-bound (~49 GFLOP vs ~20 MB effective traffic) -> route all GEMMs
// through v_wmma_f32_16x16x32_bf16. Weights pre-converted to transposed bf16
// (K-contiguous rows) so B fragments are two 16B loads/lane; activations are
// staged in LDS row-major so A fragments are two ds_load_b128 per lane.
// H is pre-converted to bf16 once so edge/node staging is a pure 16B/thread
// copy issued as global_load_async_to_lds_b128 (ASYNCcnt path).
// ---------------------------------------------------------------------------

#define D_MODEL 512
#define D_EDGE  128
#define BB      2
#define NN      2048
#define EE      65536
#define KE      1034     // 2*D_MODEL + 2 + 8
#define KEP     1056     // padded to 33*32
#define KN      640      // D_MODEL + D_EDGE
#define HID     1024     // 2*D_MODEL

typedef __attribute__((ext_vector_type(16))) __bf16        v16bf;
typedef __attribute__((ext_vector_type(8)))  float         v8f;
typedef __attribute__((ext_vector_type(4)))  unsigned int  v4u;

union FragU { v16bf v; v4u q[2]; };
union Seg8  { v4u q; unsigned short u[8]; };

__device__ __forceinline__ unsigned short f2bf(float f) {
  unsigned u = __float_as_uint(f);
  u += 0x7FFFu + ((u >> 16) & 1u);           // round-to-nearest-even
  return (unsigned short)(u >> 16);
}
__device__ __forceinline__ float siluf(float x) { return x / (1.0f + __expf(-x)); }
__device__ __forceinline__ float sigmf(float x) { return 1.0f / (1.0f + __expf(-x)); }
__device__ __forceinline__ v8f zero8() {
  v8f z;
#pragma unroll
  for (int i = 0; i < 8; ++i) z[i] = 0.0f;
  return z;
}

// A fragment (16x32 bf16): lane<16 holds row m0+lane, K [k0+8h,+8) & [k0+16+8h,+8)
__device__ __forceinline__ v16bf load_a(const unsigned short* base, int ldk,
                                        int m0, int k0, int lane) {
  const int hlf = lane >> 4;
  const unsigned short* p = base + (size_t)(m0 + (lane & 15)) * ldk + k0 + 8 * hlf;
  FragU f;
  f.q[0] = *(const v4u*)(p);
  f.q[1] = *(const v4u*)(p + 16);
  return f.v;
}
// B fragment (32x16 bf16) from transposed weights wT[n][k] (K contiguous):
// lane -> column n0+(lane&15); K [k0+8h,+8) & [k0+16+8h,+8), h = lane>>4
__device__ __forceinline__ v16bf load_b(const unsigned short* wT, int ldk,
                                        int n0, int k0, int lane) {
  const unsigned short* p = wT + (size_t)(n0 + (lane & 15)) * ldk + k0 + 8 * (lane >> 4);
  FragU f;
  f.q[0] = *(const v4u*)(p);
  f.q[1] = *(const v4u*)(p + 16);
  return f.v;
}
__device__ __forceinline__ v8f wmma_bf16(v16bf a, v16bf b, v8f c) {
  return __builtin_amdgcn_wmma_f32_16x16x32_bf16(false, a, false, b, (short)0, c,
                                                 false, false);
}

// Async 16B global->LDS copy (CDNA5 GLOBAL_LOAD_ASYNC_TO_LDS_B128, ASYNCcnt).
__device__ __forceinline__ void async_copy_b128(unsigned lds_addr,
                                                unsigned long long gaddr) {
  asm volatile("global_load_async_to_lds_b128 %0, %1, off"
               :: "v"(lds_addr), "v"(gaddr) : "memory");
}
__device__ __forceinline__ void wait_async0() {
  asm volatile("s_wait_asynccnt 0x0" ::: "memory");
}

// ---------------------------------------------------------------------------
// Kernel 1: weights -> transposed padded bf16; H -> bf16 mirror.
// ---------------------------------------------------------------------------
__global__ void convert_kernel(const float* __restrict__ We1,
                               const float* __restrict__ We2,
                               const float* __restrict__ Wn1,
                               const float* __restrict__ Wn2,
                               const float* __restrict__ H,
                               unsigned short* __restrict__ We1T,
                               unsigned short* __restrict__ We2T,
                               unsigned short* __restrict__ Wn1T,
                               unsigned short* __restrict__ Wn2T,
                               unsigned short* __restrict__ Hbf) {
  const int n1 = D_EDGE * KEP;
  const int n2 = D_EDGE * D_EDGE;
  const int n3 = HID * KN;
  const int n4 = D_MODEL * HID;
  const int n5 = BB * NN * D_MODEL;
  const int total = n1 + n2 + n3 + n4 + n5;
  for (int i = blockIdx.x * blockDim.x + threadIdx.x; i < total;
       i += gridDim.x * blockDim.x) {
    if (i < n1) {
      int n = i / KEP, k = i % KEP;
      We1T[i] = (k < KE) ? f2bf(We1[(size_t)k * D_EDGE + n]) : (unsigned short)0;
    } else if (i < n1 + n2) {
      int j = i - n1; int n = j / D_EDGE, k = j % D_EDGE;
      We2T[j] = f2bf(We2[(size_t)k * D_EDGE + n]);
    } else if (i < n1 + n2 + n3) {
      int j = i - n1 - n2; int n = j / KN, k = j % KN;
      Wn1T[j] = f2bf(Wn1[(size_t)k * HID + n]);
    } else if (i < n1 + n2 + n3 + n4) {
      int j = i - n1 - n2 - n3; int n = j / HID, k = j % HID;
      Wn2T[j] = f2bf(Wn2[(size_t)k * D_MODEL + n]);
    } else {
      int j = i - n1 - n2 - n3 - n4;
      Hbf[j] = f2bf(H[j]);
    }
  }
}

// ---------------------------------------------------------------------------
// Kernel 2: edge MLP + gate + scatter-add.  64 edges / block, 8 waves.
// ---------------------------------------------------------------------------
__global__ void __launch_bounds__(256) edge_kernel(
    const unsigned short* __restrict__ Hbf, const int* __restrict__ eidx,
    const float* __restrict__ dist2, const float* __restrict__ dlt,
    const float* __restrict__ estruct,
    const unsigned short* __restrict__ We1T, const float* __restrict__ be1,
    const unsigned short* __restrict__ We2T, const float* __restrict__ be2,
    const float* __restrict__ Wg1, const float* __restrict__ bg1,
    const float* __restrict__ Wg2, const float* __restrict__ bg2,
    float* __restrict__ agg) {
  __shared__ __align__(16) unsigned short sAk[64 * 32];   // K-slab of e_in
  __shared__ __align__(16) unsigned short sH[64 * 128];   // hidden activations
  __shared__ int   sI[64];
  __shared__ int   sJ[64];
  __shared__ float sG[64];

  const int t    = threadIdx.x;
  const int lane = t & 31;
  const int w    = t >> 5;
  const int hlf  = lane >> 4;
  const int b    = blockIdx.y;
  const int e0   = blockIdx.x * 64;

  if (t < 64) {
    const int e = e0 + t;
    sI[t] = eidx[e];
    sJ[t] = eidx[EE + e];
    // tiny gate MLP (10 -> 32 -> 1) in f32 VALU
    float gin[10];
    gin[0] = dist2[(size_t)b * EE + e];
    gin[1] = dlt[(size_t)b * EE + e];
#pragma unroll
    for (int s = 0; s < 8; ++s) gin[2 + s] = estruct[((size_t)b * EE + e) * 8 + s];
    float acc = bg2[0];
    for (int hN = 0; hN < 32; ++hN) {
      float v = bg1[hN];
#pragma unroll
      for (int q = 0; q < 10; ++q) v += gin[q] * Wg1[q * 32 + hN];
      acc += siluf(v) * Wg2[hN];
    }
    sG[t] = sigmf(acc);
  }
  __syncthreads();

  // Per-thread gather assignment: row = t>>2 (0..63), 8-col segment = t&3.
  const int rowE = t >> 2;
  const int seg8 = (t & 3) * 8;
  const unsigned short* pHi = Hbf + ((size_t)b * NN + sI[rowE]) * D_MODEL + seg8;
  const unsigned short* pHj = Hbf + ((size_t)b * NN + sJ[rowE]) * D_MODEL + seg8;
  unsigned short* ldsDstP = sAk + rowE * 32 + seg8;
  const unsigned ldsDst = (unsigned)(uintptr_t)ldsDstP;

  // GEMM1: e_in(64 x 1056) @ We1 -> h(64 x 128); wave w owns N-tile w.
  v8f acc1[4];
#pragma unroll
  for (int mt = 0; mt < 4; ++mt) acc1[mt] = zero8();

  auto gemm_step = [&](int k0) {
    v16bf bfrag = load_b(We1T, KEP, w * 16, k0, lane);
    if (k0 + 32 < KEP)
      __builtin_prefetch(We1T + (size_t)(w * 16 + (lane & 15)) * KEP + k0 + 32, 0, 3);
#pragma unroll
    for (int mt = 0; mt < 4; ++mt) {
      v16bf afrag = load_a(sAk, 32, mt * 16, 0, lane);
      acc1[mt] = wmma_bf16(afrag, bfrag, acc1[mt]);
    }
  };

  for (int kt = 0; kt < 16; ++kt) {      // Hi slabs: K [0,512)
    async_copy_b128(ldsDst, (unsigned long long)(uintptr_t)(pHi + kt * 32));
    wait_async0();
    __syncthreads();
    gemm_step(kt * 32);
    __syncthreads();
  }
  for (int kt = 0; kt < 16; ++kt) {      // Hj slabs: K [512,1024)
    async_copy_b128(ldsDst, (unsigned long long)(uintptr_t)(pHj + kt * 32));
    wait_async0();
    __syncthreads();
    gemm_step(512 + kt * 32);
    __syncthreads();
  }
  {                                      // tail: geom(2) + struct(8) + zero pad
    Seg8 tv;
#pragma unroll
    for (int q = 0; q < 8; ++q) {
      const int c = seg8 + q;            // column within 32-wide tail tile
      float v;
      if (c == 0)      v = dist2[(size_t)b * EE + e0 + rowE];
      else if (c == 1) v = dlt[(size_t)b * EE + e0 + rowE];
      else if (c < 10) v = estruct[((size_t)b * EE + e0 + rowE) * 8 + (c - 2)];
      else             v = 0.0f;
      tv.u[q] = f2bf(v);
    }
    *(v4u*)ldsDstP = tv.q;
    __syncthreads();
    gemm_step(1024);
  }

  { // h = silu(acc1 + be1) -> sH (bf16)
    const int n = w * 16 + (lane & 15);
    const float bias = be1[n];
#pragma unroll
    for (int mt = 0; mt < 4; ++mt)
#pragma unroll
      for (int r = 0; r < 8; ++r) {
        const int m = mt * 16 + r + 8 * hlf;
        sH[m * 128 + n] = f2bf(siluf(acc1[mt][r] + bias));
      }
  }
  __syncthreads();

  // GEMM2: h(64x128) @ We2 -> e_msg(64x128)
  v8f acc2[4];
#pragma unroll
  for (int mt = 0; mt < 4; ++mt) acc2[mt] = zero8();
#pragma unroll
  for (int kk = 0; kk < 4; ++kk) {
    v16bf bfrag = load_b(We2T, 128, w * 16, kk * 32, lane);
#pragma unroll
    for (int mt = 0; mt < 4; ++mt) {
      v16bf afrag = load_a(sH, 128, mt * 16, kk * 32, lane);
      acc2[mt] = wmma_bf16(afrag, bfrag, acc2[mt]);
    }
  }

  { // gate + scatter-add into agg (f32 global atomics, reference semantics)
    const int n = w * 16 + (lane & 15);
    const float bias = be2[n];
#pragma unroll
    for (int mt = 0; mt < 4; ++mt)
#pragma unroll
      for (int r = 0; r < 8; ++r) {
        const int m = mt * 16 + r + 8 * hlf;
        const float v = siluf(acc2[mt][r] + bias) * sG[m];
        atomicAdd(&agg[((size_t)b * NN + sI[m]) * D_EDGE + n], v);
      }
  }
}

// ---------------------------------------------------------------------------
// Kernel 3: node MLP + residual + LayerNorm.  32 rows / block, 8 waves.
// Intermediate 1024 processed in 8 chunks of 128; LDS kept at exactly 64 KB.
// ---------------------------------------------------------------------------
__global__ void __launch_bounds__(256) node_kernel(
    const float* __restrict__ H, const unsigned short* __restrict__ Hbf,
    const float* __restrict__ agg,
    const unsigned short* __restrict__ Wn1T, const float* __restrict__ bn1,
    const unsigned short* __restrict__ Wn2T, const float* __restrict__ bn2,
    const float* __restrict__ ln_g, const float* __restrict__ ln_b,
    float* __restrict__ out) {
  __shared__ __align__(16) unsigned char nsmem[65536];
  unsigned short* sA  = (unsigned short*)nsmem;            // 32 x 640 bf16 (40 KB)
  unsigned short* sH1 = (unsigned short*)(nsmem + 40960);  // 32 x 128 bf16 ( 8 KB)
  float*          sX  = (float*)nsmem;                     // 32 x 512 f32  (64 KB, aliases)

  const int t    = threadIdx.x;
  const int lane = t & 31;
  const int w    = t >> 5;
  const int hlf  = lane >> 4;
  const size_t r0 = (size_t)blockIdx.x * 32;               // flat row over B*N

  // stage n_in = [Hbf | bf16(agg)] as 8-element segments (2560 segs / block);
  // the Hbf half goes through the async-LDS path, agg half is converted inline.
  for (int sidx = t; sidx < 32 * (KN / 8); sidx += 256) {
    const int row = sidx / (KN / 8);
    const int c0  = (sidx % (KN / 8)) * 8;
    unsigned short* dst = sA + row * KN + c0;
    if (c0 < D_MODEL) {
      async_copy_b128((unsigned)(uintptr_t)dst,
                      (unsigned long long)(uintptr_t)(Hbf + (r0 + row) * D_MODEL + c0));
    } else {
      const float* pa = agg + (r0 + row) * D_EDGE + (c0 - D_MODEL);
      Seg8 tv;
#pragma unroll
      for (int q = 0; q < 8; ++q) tv.u[q] = f2bf(pa[q]);
      *(v4u*)dst = tv.q;
    }
  }
  wait_async0();

  const int mt2 = w & 1;            // this wave's M-tile in GEMM-B
  const int ng0 = (w >> 1) * 8;     // this wave's N-tile group base (8 tiles)
  v8f accU[8];
#pragma unroll
  for (int g = 0; g < 8; ++g) accU[g] = zero8();

  for (int c = 0; c < 8; ++c) {     // chunks of the 1024-wide intermediate
    __syncthreads();
    // GEMM-A: h1c = silu(n_in @ Wn1[:, c*128 + w*16 ..]); wave owns N-tile w
    v8f a1[2]; a1[0] = zero8(); a1[1] = zero8();
    const int ncol = c * 128 + w * 16;
    for (int kk = 0; kk < KN / 32; ++kk) {
      v16bf bfrag = load_b(Wn1T, KN, ncol, kk * 32, lane);
      v16bf af0 = load_a(sA, KN, 0, kk * 32, lane);
      v16bf af1 = load_a(sA, KN, 16, kk * 32, lane);
      a1[0] = wmma_bf16(af0, bfrag, a1[0]);
      a1[1] = wmma_bf16(af1, bfrag, a1[1]);
    }
    {
      const int n = w * 16 + (lane & 15);
      const float bias = bn1[c * 128 + n];
#pragma unroll
      for (int mt = 0; mt < 2; ++mt)
#pragma unroll
        for (int r = 0; r < 8; ++r) {
          const int m = mt * 16 + r + 8 * hlf;
          sH1[m * 128 + n] = f2bf(siluf(a1[mt][r] + bias));
        }
    }
    __syncthreads();
    // GEMM-B partial: upd += h1c @ Wn2[c*128 .., :]
#pragma unroll
    for (int kk = 0; kk < 4; ++kk) {
      v16bf afrag = load_a(sH1, 128, mt2 * 16, kk * 32, lane);
#pragma unroll
      for (int g = 0; g < 8; ++g) {
        v16bf bfrag = load_b(Wn2T, HID, (ng0 + g) * 16, c * 128 + kk * 32, lane);
        accU[g] = wmma_bf16(afrag, bfrag, accU[g]);
      }
    }
  }
  __syncthreads();                  // sA/sH1 dead; sX may now alias them

  { // x = H + upd + bn2 -> sX  (f32 residual precision from original H)
#pragma unroll
    for (int g = 0; g < 8; ++g) {
      const int n = (ng0 + g) * 16 + (lane & 15);
      const float bias = bn2[n];
#pragma unroll
      for (int r = 0; r < 8; ++r) {
        const int m = mt2 * 16 + r + 8 * hlf;
        sX[m * D_MODEL + n] = accU[g][r] + bias + H[(r0 + m) * D_MODEL + n];
      }
    }
  }
  __syncthreads();

  // LayerNorm: wave w handles rows w*4 .. w*4+3 (wave32 shuffle reduction)
  for (int rr = 0; rr < 4; ++rr) {
    const int row = w * 4 + rr;
    float s = 0.0f, sq = 0.0f;
#pragma unroll
    for (int q = 0; q < 16; ++q) {
      const float v = sX[row * D_MODEL + lane + q * 32];
      s += v; sq += v * v;
    }
#pragma unroll
    for (int off = 16; off >= 1; off >>= 1) {
      s  += __shfl_xor(s, off, 32);
      sq += __shfl_xor(sq, off, 32);
    }
    const float mu   = s * (1.0f / D_MODEL);
    const float var  = sq * (1.0f / D_MODEL) - mu * mu;
    const float rstd = rsqrtf(var + 1e-5f);
#pragma unroll
    for (int q = 0; q < 16; ++q) {
      const int n = lane + q * 32;
      out[(r0 + row) * D_MODEL + n] =
          (sX[row * D_MODEL + n] - mu) * rstd * ln_g[n] + ln_b[n];
    }
  }
}

// ---------------------------------------------------------------------------
extern "C" void kernel_launch(void* const* d_in, const int* in_sizes, int n_in,
                              void* d_out, int out_size, void* d_ws, size_t ws_size,
                              hipStream_t stream) {
  (void)in_sizes; (void)n_in; (void)out_size; (void)ws_size;
  const float* H       = (const float*)d_in[0];
  const int*   eidx    = (const int*)  d_in[1];
  const float* dist2   = (const float*)d_in[2];
  const float* dlt     = (const float*)d_in[3];
  const float* estruct = (const float*)d_in[4];
  const float* We1 = (const float*)d_in[5];
  const float* be1 = (const float*)d_in[6];
  const float* We2 = (const float*)d_in[7];
  const float* be2 = (const float*)d_in[8];
  const float* Wg1 = (const float*)d_in[9];
  const float* bg1 = (const float*)d_in[10];
  const float* Wg2 = (const float*)d_in[11];
  const float* bg2 = (const float*)d_in[12];
  const float* Wn1 = (const float*)d_in[13];
  const float* bn1 = (const float*)d_in[14];
  const float* Wn2 = (const float*)d_in[15];
  const float* bn2 = (const float*)d_in[16];
  const float* lng = (const float*)d_in[17];
  const float* lnb = (const float*)d_in[18];

  char* ws = (char*)d_ws;
  size_t off = 0;
  auto take = [&](size_t bytes) -> void* {
    void* p = ws + off;
    off = (off + bytes + 255) & ~(size_t)255;
    return p;
  };
  unsigned short* We1T = (unsigned short*)take((size_t)D_EDGE * KEP * 2);
  unsigned short* We2T = (unsigned short*)take((size_t)D_EDGE * D_EDGE * 2);
  unsigned short* Wn1T = (unsigned short*)take((size_t)HID * KN * 2);
  unsigned short* Wn2T = (unsigned short*)take((size_t)D_MODEL * HID * 2);
  unsigned short* Hbf  = (unsigned short*)take((size_t)BB * NN * D_MODEL * 2);
  float* agg = (float*)take((size_t)BB * NN * D_EDGE * 4);

  (void)hipMemsetAsync(agg, 0, (size_t)BB * NN * D_EDGE * 4, stream);
  convert_kernel<<<1024, 256, 0, stream>>>(We1, We2, Wn1, Wn2, H,
                                           We1T, We2T, Wn1T, Wn2T, Hbf);
  dim3 egrid(EE / 64, BB);
  edge_kernel<<<egrid, 256, 0, stream>>>(Hbf, eidx, dist2, dlt, estruct,
                                         We1T, be1, We2T, be2,
                                         Wg1, bg1, Wg2, bg2, agg);
  node_kernel<<<(BB * NN) / 32, 256, 0, stream>>>(H, Hbf, agg, Wn1T, bn1,
                                                  Wn2T, bn2, lng, lnb,
                                                  (float*)d_out);
}
